// GNN_82068235092795
// MI455X (gfx1250) — matrix operand order repaired and checked by reading, not compiled, not run on hardware
//
#include <hip/hip_runtime.h>
#include <hip/hip_bf16.h>
#include <math.h>

#define B_  128
#define T_  512
#define N_  17
#define E_  32
#define C_  16
#define H_  256
#define G_  (B_ * T_)     // 65536 graphs / tokens
#define K0P 288           // N*C=272 padded to multiple of 32
#define K12 512
#define NG  1024          // 4*H gate width
#define HPAD 264          // LDS row stride (ushorts) for h, bank-spread

// dynamic LDS for scan: h (32 x HPAD) + double-buffered whh chunk (2 x 1024 x 32)
#define SCAN_LDS_USHORTS (32 * HPAD + 2 * NG * 32)

typedef __attribute__((ext_vector_type(16))) __bf16 v16bf;
typedef __attribute__((ext_vector_type(8)))  __bf16 v8bf;
typedef __attribute__((ext_vector_type(8)))  float  v8f;

union F16U { v16bf v; v8bf h[2]; unsigned short s[16]; };
union F8U  { v8bf v; unsigned short s[8]; };

__device__ __forceinline__ unsigned short f2bf(float f) {
  unsigned u = __float_as_uint(f);
  unsigned r = u + 0x7FFFu + ((u >> 16) & 1u);   // round-to-nearest-even
  return (unsigned short)(r >> 16);
}
__device__ __forceinline__ float bf2f(unsigned short h) {
  return __uint_as_float(((unsigned)h) << 16);
}
__device__ __forceinline__ float sigmoidf_(float x) { return 1.0f / (1.0f + __expf(-x)); }

// ---------------------------------------------------------------------------
// Weight convert fp32 -> bf16 with optional K padding
// ---------------------------------------------------------------------------
__global__ void cvt_pad_kernel(const float* __restrict__ src,
                               unsigned short* __restrict__ dst,
                               long rows, int ks, int kd) {
  long i = (long)blockIdx.x * blockDim.x + threadIdx.x;
  long tot = rows * kd;
  if (i >= tot) return;
  long r = i / kd;
  int  k = (int)(i - r * kd);
  dst[i] = (k < ks) ? f2bf(src[r * ks + k]) : (unsigned short)0;
}

// ---------------------------------------------------------------------------
// GIN: 3 graph-conv layers fused, 8 graphs per block, one node per lane.
// Writes bf16 x_time (G x 288), cols 272..287 zero-padded.
// ---------------------------------------------------------------------------
__global__ __launch_bounds__(256)
void gin_kernel(const float* __restrict__ x, const int* __restrict__ ei,
                const float* __restrict__ g0w1, const float* __restrict__ g0b1,
                const float* __restrict__ g0w2, const float* __restrict__ g0b2,
                const float* __restrict__ g12w1, const float* __restrict__ g12b1,
                const float* __restrict__ g12w2, const float* __restrict__ g12b2,
                unsigned short* __restrict__ xt) {
  __shared__ int   es[8][2][E_];
  __shared__ float xs[8][N_][2];
  __shared__ float hsh[8][N_][C_];
  const int tx = threadIdx.x;           // 0..31 (node lane)
  const int gg = threadIdx.y;           // 0..7  (graph in block)
  const long g = (long)blockIdx.x * 8 + gg;

  ((int*)es[gg])[tx]      = ei[g * 64 + tx];
  ((int*)es[gg])[tx + 32] = ei[g * 64 + 32 + tx];
  if (tx < N_) {
    xs[gg][tx][0] = x[g * (N_ * 2) + tx * 2 + 0];
    xs[gg][tx][1] = x[g * (N_ * 2) + tx * 2 + 1];
  }
  __syncthreads();

  float h[C_];
  for (int j = 0; j < C_; ++j) h[j] = 0.0f;

  // ---- layer 0: feat dim 2 -> 16 ----
  if (tx < N_) {
    float a0 = 0.f, a1 = 0.f;
    for (int e = 0; e < E_; ++e) {
      if (es[gg][1][e] == tx) { int s = es[gg][0][e]; a0 += xs[gg][s][0]; a1 += xs[gg][s][1]; }
    }
    float u0 = xs[gg][tx][0] + a0, u1 = xs[gg][tx][1] + a1;
    float t1[C_];
    for (int j = 0; j < C_; ++j) {
      float v = g0b1[j] + g0w1[j * 2] * u0 + g0w1[j * 2 + 1] * u1;
      t1[j] = fmaxf(v, 0.f);
    }
    for (int j = 0; j < C_; ++j) {
      float v = g0b2[j];
      for (int k = 0; k < C_; ++k) v += t1[k] * g0w2[j * C_ + k];
      h[j] = fmaxf(v, 0.f);                 // inter-layer relu (layer 0)
    }
  }

  // ---- layers 1,2: 16 -> 16 ----
  for (int li = 0; li < 2; ++li) {
    __syncthreads();
    if (tx < N_) for (int j = 0; j < C_; ++j) hsh[gg][tx][j] = h[j];
    __syncthreads();
    if (tx < N_) {
      float agg[C_];
      for (int j = 0; j < C_; ++j) agg[j] = 0.f;
      for (int e = 0; e < E_; ++e) {
        if (es[gg][1][e] == tx) {
          int s = es[gg][0][e];
          for (int j = 0; j < C_; ++j) agg[j] += hsh[gg][s][j];
        }
      }
      const float* w1 = g12w1 + li * C_ * C_;
      const float* b1 = g12b1 + li * C_;
      const float* w2 = g12w2 + li * C_ * C_;
      const float* b2 = g12b2 + li * C_;
      float u[C_], t1[C_];
      for (int j = 0; j < C_; ++j) u[j] = h[j] + agg[j];
      for (int j = 0; j < C_; ++j) {
        float v = b1[j];
        for (int k = 0; k < C_; ++k) v += u[k] * w1[j * C_ + k];
        t1[j] = fmaxf(v, 0.f);
      }
      for (int j = 0; j < C_; ++j) {
        float v = b2[j];
        for (int k = 0; k < C_; ++k) v += t1[k] * w2[j * C_ + k];
        h[j] = (li == 0) ? fmaxf(v, 0.f) : v;  // relu only between layers
      }
    }
  }

  if (tx < N_) {
    for (int j = 0; j < C_; ++j) xt[g * K0P + tx * C_ + j] = f2bf(h[j]);
  } else if (tx == N_) {
    for (int j = 0; j < 16; ++j) xt[g * K0P + 272 + j] = 0;
  }
}

// ---------------------------------------------------------------------------
// Input projection. A: (G x K) bf16 token-major; W: (1024 x K) bf16.
// One wave = 16M x 128N strip of 16x16x32 bf16 WMMAs.
// Output XP is stored TRANSPOSED for the scan: XP[t][n][b] so the scan can
// read 8 consecutive batch rows with one 16B load.
// ---------------------------------------------------------------------------
__global__ __launch_bounds__(256)
void proj_kernel(const unsigned short* __restrict__ A,
                 const unsigned short* __restrict__ W,
                 const float* __restrict__ bih, const float* __restrict__ bhh,
                 unsigned short* __restrict__ XP, int K) {
  const int lane = threadIdx.x & 31;
  const int wv   = threadIdx.x >> 5;      // 0..7
  const int l15  = lane & 15;
  const int hi   = lane >> 4;             // 0/1
  const long m   = (long)blockIdx.x * 16 + l15;
  const unsigned short* arow = A + m * K;
  const int nbase = wv * 128;

  v8f acc[8];
  for (int j = 0; j < 8; ++j) {
    int n = nbase + j * 16 + l15;
    float bv = bih[n] + bhh[n];
    for (int r = 0; r < 8; ++r) acc[j][r] = bv;
  }

  const int KC = K >> 5;
  for (int kc = 0; kc < KC; ++kc) {
    F16U af;
    const unsigned short* ap = arow + kc * 32 + (hi ? 8 : 0);
    __builtin_prefetch(arow + (kc + 2) * 32, 0, 3);
    af.h[0] = *(const v8bf*)(ap);
    af.h[1] = *(const v8bf*)(ap + 16);
    for (int j = 0; j < 8; ++j) {
      int n = nbase + j * 16 + l15;
      const unsigned short* wp = W + (long)n * K + kc * 32 + (hi ? 8 : 0);
      F16U bf; bf.h[0] = *(const v8bf*)wp; bf.h[1] = *(const v8bf*)(wp + 16);
      acc[j] = __builtin_amdgcn_wmma_f32_16x16x32_bf16(
          false, af.v, false, bf.v, (short)0, acc[j], false, false);
    }
  }

  // 16-row tile never crosses a batch boundary (T=512 is a multiple of 16)
  const int bidx  = (blockIdx.x * 16) / T_;
  const int tbase = (blockIdx.x * 16) % T_ + hi * 8;
  for (int j = 0; j < 8; ++j) {
    int n = nbase + j * 16 + l15;
    for (int r = 0; r < 8; ++r)
      XP[((long)(tbase + r) * NG + n) * B_ + bidx] = f2bf(acc[j][r]);
  }
}

// ---------------------------------------------------------------------------
// LSTM scan. grid = (4 batch-slices, 2 directions). Block = 512 thr = 16 waves.
// Wave w owns hidden-tile hj=w (16 cols) x 2 M-tiles (32 batch rows).
// h kept in LDS (bf16), c in registers. whh is double-buffered into LDS with
// global_load_async_to_lds_b128: each wave stages exactly the 64 rows it will
// read, so chunk kc+1 streams in (ASYNCcnt) while chunk kc is consumed by
// WMMAs -- no extra barriers needed.
// ---------------------------------------------------------------------------
__global__ __launch_bounds__(512)
void scan_kernel(const unsigned short* __restrict__ XP,
                 const unsigned short* __restrict__ WHH,
                 unsigned short* __restrict__ HS,
                 long xpDirStride) {
  const int dir = blockIdx.y;
  const int b0  = blockIdx.x * 32;
  const unsigned short* xp  = XP + (long)dir * xpDirStride;
  const unsigned short* whh = WHH + (long)dir * ((long)NG * H_);
  const int lane = threadIdx.x & 31;
  const int wv   = threadIdx.x >> 5;      // 0..15 = hidden tile
  const int l15  = lane & 15;
  const int hi   = lane >> 4;

  extern __shared__ unsigned short smem[];
  unsigned short* hlds  = smem;                 // 32 x HPAD
  unsigned short* wpool = smem + 32 * HPAD;     // 2 x (1024 x 32)

  for (int i = threadIdx.x; i < 32 * HPAD; i += 512) hlds[i] = 0;

  // stage this wave's 64 rows (4 gate tiles x 16) of a 32-wide K chunk of whh
  auto stage_chunk = [&](int kc, int dsel) {
    unsigned short* wb = wpool + dsel * (NG * 32);
    const unsigned short* gbase = whh + kc * 32;
#pragma unroll
    for (int it = 0; it < 8; ++it) {
      int u   = it * 32 + lane;                 // 0..255 : 16B unit
      int row = u >> 2;                         // 0..63  = gi*16 + r
      int sub = u & 3;
      int n   = ((row >> 4) << 8) + wv * 16 + (row & 15);
      const unsigned short* gp = gbase + (long)n * H_ + sub * 8;
      unsigned short* lp = wb + (wv * 64 + row) * 32 + sub * 8;
      unsigned ldso = (unsigned)(unsigned long long)(void*)lp;
      asm volatile("global_load_async_to_lds_b128 %0, %1, off"
                   :: "v"(ldso), "v"(gp) : "memory");
    }
  };

  v8f cst[2];
  for (int mt = 0; mt < 2; ++mt)
    for (int r = 0; r < 8; ++r) cst[mt][r] = 0.f;
  __syncthreads();

  stage_chunk(0, 0);

  for (int t = 0; t < T_; ++t) {
    const int tt = dir ? (T_ - 1 - t) : t;
    v8f acc[4][2];
    // z = xp_t  (biases folded in by proj); XP layout [t][n][b]
    for (int gi = 0; gi < 4; ++gi) {
      int n = gi * 256 + wv * 16 + l15;
      const unsigned short* xr = xp + ((long)tt * NG + n) * B_ + b0;
      for (int mt = 0; mt < 2; ++mt) {
        F8U xu;
        xu.v = *(const v8bf*)(xr + mt * 16 + hi * 8);
        for (int r = 0; r < 8; ++r) acc[gi][mt][r] = bf2f(xu.s[r]);
      }
    }
    // z += h_{t-1} @ whh^T  (whh chunks double-buffered via ASYNCcnt)
    for (int kc = 0; kc < 8; ++kc) {
      stage_chunk((kc + 1) & 7, (kc + 1) & 1);
      asm volatile("s_wait_asynccnt 0x8" ::: "memory");   // chunk kc resident
      const unsigned short* wb = wpool + (kc & 1) * (NG * 32);
      F16U af[2];
      for (int mt = 0; mt < 2; ++mt) {
        const unsigned short* ap = &hlds[(mt * 16 + l15) * HPAD + kc * 32 + (hi ? 8 : 0)];
        af[mt].h[0] = *(const v8bf*)ap;
        af[mt].h[1] = *(const v8bf*)(ap + 16);
      }
      for (int gi = 0; gi < 4; ++gi) {
        const unsigned short* bp = wb + (wv * 64 + gi * 16 + l15) * 32 + (hi ? 8 : 0);
        F16U bf; bf.h[0] = *(const v8bf*)bp; bf.h[1] = *(const v8bf*)(bp + 16);
        for (int mt = 0; mt < 2; ++mt)
          acc[gi][mt] = __builtin_amdgcn_wmma_f32_16x16x32_bf16(
              false, af[mt].v, false, bf.v, (short)0, acc[gi][mt], false, false);
      }
    }
    __syncthreads();   // all waves done reading h_{t-1} from LDS

    // gates + state update; lane mapping identical across the 4 gate tiles
    for (int mt = 0; mt < 2; ++mt) {
      for (int r = 0; r < 8; ++r) {
        float iv = sigmoidf_(acc[0][mt][r]);
        float fv = sigmoidf_(acc[1][mt][r]);
        float gv = tanhf(acc[2][mt][r]);
        float ov = sigmoidf_(acc[3][mt][r]);
        float c  = fv * cst[mt][r] + iv * gv;
        cst[mt][r] = c;
        float hv = ov * tanhf(c);
        int col = wv * 16 + l15;            // hidden unit 0..255
        int lm  = mt * 16 + hi * 8 + r;     // local batch row 0..31
        unsigned short hb = f2bf(hv);
        hlds[lm * HPAD + col] = hb;
        long bb = b0 + lm;
        HS[((bb * (long)T_) + tt) * 512 + dir * 256 + col] = hb;
      }
    }
    __syncthreads();   // h_t fully written before next step reads
  }
}

// ---------------------------------------------------------------------------
// Attention pooling: softmax over T of (hs @ attn_w + b), weighted sum -> xv
// ---------------------------------------------------------------------------
__global__ __launch_bounds__(256)
void attn_kernel(const unsigned short* __restrict__ HS,
                 const float* __restrict__ aw, const float* __restrict__ ab,
                 float* __restrict__ XV) {
  const int b = blockIdx.x;
  __shared__ float lg[T_];
  __shared__ float red[256];
  for (int t = threadIdx.x; t < T_; t += 256) {
    const unsigned short* row = HS + ((long)b * T_ + t) * 512;
    float s = ab[0];
    for (int j = 0; j < 512; ++j) s += bf2f(row[j]) * aw[j];
    lg[t] = s;
  }
  __syncthreads();
  float mx = -1e30f;
  for (int t = threadIdx.x; t < T_; t += 256) mx = fmaxf(mx, lg[t]);
  red[threadIdx.x] = mx; __syncthreads();
  for (int s = 128; s > 0; s >>= 1) {
    if (threadIdx.x < s) red[threadIdx.x] = fmaxf(red[threadIdx.x], red[threadIdx.x + s]);
    __syncthreads();
  }
  mx = red[0]; __syncthreads();
  float sm = 0.f;
  for (int t = threadIdx.x; t < T_; t += 256) {
    float e = __expf(lg[t] - mx); lg[t] = e; sm += e;
  }
  red[threadIdx.x] = sm; __syncthreads();
  for (int s = 128; s > 0; s >>= 1) {
    if (threadIdx.x < s) red[threadIdx.x] += red[threadIdx.x + s];
    __syncthreads();
  }
  float inv = 1.0f / red[0];
  __syncthreads();
  for (int j = threadIdx.x; j < 512; j += 256) {
    float s = 0.f;
    for (int t = 0; t < T_; ++t)
      s += lg[t] * bf2f(HS[((long)b * T_ + t) * 512 + j]);
    XV[b * 512 + j] = s * inv;
  }
}

// ---------------------------------------------------------------------------
// Small FC / batchnorm(+relu) / heads
// ---------------------------------------------------------------------------
__global__ void fc_kernel(const float* __restrict__ X, const float* __restrict__ W,
                          const float* __restrict__ bias, float* __restrict__ Y,
                          int rows, int K, int Ncols) {
  int i = blockIdx.x * blockDim.x + threadIdx.x;
  if (i >= rows * Ncols) return;
  int r = i / Ncols, j = i - r * Ncols;
  float s = bias[j];
  for (int k = 0; k < K; ++k) s += X[r * K + k] * W[j * K + k];
  Y[i] = s;
}

__global__ __launch_bounds__(128)
void bn_relu_kernel(const float* __restrict__ Y, const float* __restrict__ g,
                    const float* __restrict__ bb, float* __restrict__ O, int Ncols) {
  const int j = blockIdx.x, r = threadIdx.x;   // 128 rows
  __shared__ float red[128];
  float v = Y[r * Ncols + j];
  red[r] = v; __syncthreads();
  for (int s = 64; s > 0; s >>= 1) {
    if (r < s) red[r] += red[r + s];
    __syncthreads();
  }
  float mean = red[0] * (1.0f / 128.0f); __syncthreads();
  float d = v - mean;
  red[r] = d * d; __syncthreads();
  for (int s = 64; s > 0; s >>= 1) {
    if (r < s) red[r] += red[r + s];
    __syncthreads();
  }
  float var = red[0] * (1.0f / 128.0f);
  float o = d * rsqrtf(var + 1e-5f) * g[j] + bb[j];
  O[r * Ncols + j] = fmaxf(o, 0.f);
}

__global__ void heads_kernel(const float* __restrict__ YR,
                             const float* __restrict__ iw, const float* __restrict__ ib,
                             const float* __restrict__ aw, const float* __restrict__ ab,
                             const float* __restrict__ cw, const float* __restrict__ cb,
                             float* __restrict__ out) {
  const int b = blockIdx.x, j = threadIdx.x;   // 16 threads
  const float* xr = YR + b * 16;
  if (j < 3) {
    float s = ib[j]; for (int k = 0; k < 16; ++k) s += xr[k] * iw[j * 16 + k];
    out[b * 3 + j] = s;
    float s2 = ab[j]; for (int k = 0; k < 16; ++k) s2 += xr[k] * aw[j * 16 + k];
    out[384 + b * 3 + j] = s2;
  }
  if (j < 10) {
    float s3 = cb[j]; for (int k = 0; k < 16; ++k) s3 += xr[k] * cw[j * 16 + k];
    out[768 + b * 10 + j] = s3;
  }
}

// ---------------------------------------------------------------------------
extern "C" void kernel_launch(void* const* d_in, const int* in_sizes, int n_in,
                              void* d_out, int out_size, void* d_ws, size_t ws_size,
                              hipStream_t stream) {
  (void)in_sizes; (void)n_in; (void)out_size; (void)ws_size;
  const float* x      = (const float*)d_in[0];
  const int*   ei     = (const int*)d_in[1];
  /* d_in[2] edge_attr: unused by reference */
  const float* g0w1   = (const float*)d_in[3];
  const float* g0b1   = (const float*)d_in[4];
  const float* g0w2   = (const float*)d_in[5];
  const float* g0b2   = (const float*)d_in[6];
  const float* g12w1  = (const float*)d_in[7];
  const float* g12b1  = (const float*)d_in[8];
  const float* g12w2  = (const float*)d_in[9];
  const float* g12b2  = (const float*)d_in[10];
  const float* l0wih  = (const float*)d_in[11];
  const float* l12wih = (const float*)d_in[12];
  const float* lwhh   = (const float*)d_in[13];
  const float* lbih   = (const float*)d_in[14];
  const float* lbhh   = (const float*)d_in[15];
  const float* attn_w = (const float*)d_in[16];
  const float* attn_b = (const float*)d_in[17];
  const float* fc1_w  = (const float*)d_in[18];
  const float* fc1_b  = (const float*)d_in[19];
  const float* bn1_g  = (const float*)d_in[20];
  const float* bn1_b  = (const float*)d_in[21];
  const float* fc2_w  = (const float*)d_in[22];
  const float* fc2_b  = (const float*)d_in[23];
  const float* bn2_g  = (const float*)d_in[24];
  const float* bn2_b  = (const float*)d_in[25];
  const float* int_w  = (const float*)d_in[26];
  const float* int_b  = (const float*)d_in[27];
  const float* att_w  = (const float*)d_in[28];
  const float* att_b  = (const float*)d_in[29];
  const float* act_w  = (const float*)d_in[30];
  const float* act_b  = (const float*)d_in[31];

  char* ws = (char*)d_ws;
  size_t off = 0;
  auto alloc = [&](size_t bytes) -> void* {
    void* p = ws + off;
    off = (off + bytes + 255) & ~(size_t)255;
    return p;
  };
  unsigned short* xt    = (unsigned short*)alloc((size_t)G_ * K0P * 2);     //  ~38 MB
  unsigned short* xp    = (unsigned short*)alloc((size_t)2 * G_ * NG * 2);  // ~268 MB
  unsigned short* hsA   = (unsigned short*)alloc((size_t)G_ * 512 * 2);     //  ~67 MB
  unsigned short* hsB   = (unsigned short*)alloc((size_t)G_ * 512 * 2);     //  ~67 MB
  unsigned short* wih0  = (unsigned short*)alloc((size_t)2 * NG * K0P * 2);
  unsigned short* wih12 = (unsigned short*)alloc((size_t)4 * NG * K12 * 2);
  unsigned short* whhb  = (unsigned short*)alloc((size_t)6 * NG * H_ * 2);
  float* xv   = (float*)alloc(128 * 512 * 4);
  float* y64  = (float*)alloc(128 * 64 * 4);
  float* y64b = (float*)alloc(128 * 64 * 4);
  float* y16  = (float*)alloc(128 * 16 * 4);
  float* y16b = (float*)alloc(128 * 16 * 4);

  // ---- weight conversion to bf16 (with K padding for layer-0 wih) ----
  {
    long tot = 2L * NG * K0P;
    cvt_pad_kernel<<<(int)((tot + 255) / 256), 256, 0, stream>>>(l0wih, wih0, 2L * NG, 272, K0P);
  }
  {
    long tot = 4L * NG * K12;
    cvt_pad_kernel<<<(int)((tot + 255) / 256), 256, 0, stream>>>(l12wih, wih12, 4L * NG, K12, K12);
  }
  {
    long tot = 6L * NG * H_;
    cvt_pad_kernel<<<(int)((tot + 255) / 256), 256, 0, stream>>>(lwhh, whhb, 6L * NG, H_, H_);
  }

  // ---- GIN ----
  gin_kernel<<<G_ / 8, dim3(32, 8), 0, stream>>>(x, ei, g0w1, g0b1, g0w2, g0b2,
                                                 g12w1, g12b1, g12w2, g12b2, xt);

  // ---- 3 x bidirectional LSTM ----
  const unsigned short* Ain = xt;
  int K = K0P;
  unsigned short* hbuf[2] = { hsA, hsB };
  unsigned short* hout = hsA;
  for (int l = 0; l < 3; ++l) {
    hout = hbuf[l & 1];
    for (int dir = 0; dir < 2; ++dir) {
      const unsigned short* W = (l == 0)
          ? (wih0 + (size_t)dir * NG * K0P)
          : (wih12 + ((size_t)(l - 1) * 2 + dir) * NG * K12);
      const float* bi = lbih + ((size_t)l * 2 + dir) * NG;
      const float* bh = lbhh + ((size_t)l * 2 + dir) * NG;
      unsigned short* xpd = xp + (size_t)dir * G_ * NG;
      proj_kernel<<<G_ / 16, 256, 0, stream>>>(Ain, W, bi, bh, xpd, K);
    }
    scan_kernel<<<dim3(4, 2), 512, SCAN_LDS_USHORTS * 2, stream>>>(
        xp, whhb + (size_t)l * 2 * NG * H_, hout, (long)G_ * NG);
    Ain = hout;
    K = 512;
  }

  // ---- attention pooling + heads ----
  attn_kernel<<<128, 256, 0, stream>>>(Ain, attn_w, attn_b, xv);
  fc_kernel<<<(128 * 64 + 255) / 256, 256, 0, stream>>>(xv, fc1_w, fc1_b, y64, 128, 512, 64);
  bn_relu_kernel<<<64, 128, 0, stream>>>(y64, bn1_g, bn1_b, y64b, 64);
  fc_kernel<<<(128 * 16 + 255) / 256, 256, 0, stream>>>(y64b, fc2_w, fc2_b, y16, 128, 64, 16);
  bn_relu_kernel<<<16, 128, 0, stream>>>(y16, bn2_g, bn2_b, y16b, 16);
  heads_kernel<<<128, 16, 0, stream>>>(y16b, int_w, int_b, att_w, att_b, act_w, act_b,
                                       (float*)d_out);
}
